// XLMRobertaNERCRF_39694087749667
// MI455X (gfx1250) — matrix operand order repaired and checked by reading, not compiled, not run on hardware
//
#include <hip/hip_runtime.h>

// ---------------------------------------------------------------------------
// Problem constants (from reference)
// ---------------------------------------------------------------------------
#define B_   16
#define S_   256
#define W_   128
#define H_   768
#define L_   12
#define F_   3072
#define NH_  12
#define T_   9
#define DH_  64
#define NTOK (B_ * S_)          // 4096
#define IGNORE_IDX (-100)

typedef __bf16 bf16_t;
typedef __attribute__((ext_vector_type(16))) __bf16 v16bf;
typedef __attribute__((ext_vector_type(4)))  __bf16 v4bf;
typedef __attribute__((ext_vector_type(8)))  float  v8f;
typedef __attribute__((ext_vector_type(4)))  float  v4f;

// ---------------------------------------------------------------------------
// Helpers
// ---------------------------------------------------------------------------
__device__ __forceinline__ bf16_t to_bf16(float f) {
    union { float f; unsigned u; } in; in.f = f;
    unsigned u = in.u;
    u += 0x7FFFu + ((u >> 16) & 1u);          // round-to-nearest-even
    union { unsigned short s; bf16_t b; } out;
    out.s = (unsigned short)(u >> 16);
    return out.b;
}

// Low 32 bits of a generic pointer to LDS == wave-relative LDS byte offset.
__device__ __forceinline__ unsigned lds_off_u32(const void* p) {
    return (unsigned)reinterpret_cast<size_t>(p);
}

// CDNA5 async DMA: 16B global -> LDS, no VGPR round trip (ASYNCcnt tracked).
__device__ __forceinline__ void async_copy_b128(unsigned lds_byte_off,
                                                const void* sbase,
                                                unsigned voff_bytes) {
    asm volatile("global_load_async_to_lds_b128 %0, %1, %2"
                 :
                 : "v"(lds_byte_off), "v"(voff_bytes), "s"(sbase)
                 : "memory");
}

template <int N>
__device__ __forceinline__ void wait_asynccnt() {
    asm volatile("s_wait_asynccnt %0" : : "i"(N) : "memory");
}

// ---------------------------------------------------------------------------
// fp32 -> bf16 convert, 4 elements/thread (n % 4 == 0 for all call sites)
// ---------------------------------------------------------------------------
__global__ __launch_bounds__(256)
void cvt_bf16_kernel(const float* __restrict__ in, bf16_t* __restrict__ out, int n4) {
    int i = blockIdx.x * 256 + threadIdx.x;
    if (i < n4) {
        v4f v = reinterpret_cast<const v4f*>(in)[i];
        v4bf o;
        #pragma unroll
        for (int j = 0; j < 4; ++j) o[j] = to_bf16(v[j]);
        reinterpret_cast<v4bf*>(out)[i] = o;
    }
}

// ---------------------------------------------------------------------------
// Transpose + convert: in (K x N fp32, row-major) -> out (N x K bf16, row-major)
// K, N multiples of 32. blockDim = (32, 8). Two-phase staging so loads clause.
// ---------------------------------------------------------------------------
__global__ __launch_bounds__(256)
void transpose_bf16_kernel(const float* __restrict__ in, bf16_t* __restrict__ out,
                           int K, int N) {
    __shared__ float tile[32][33];
    int kb = blockIdx.x * 32, nb = blockIdx.y * 32;
    float r[4];
    #pragma unroll
    for (int i = 0; i < 4; ++i)
        r[i] = in[(size_t)(kb + threadIdx.y + i * 8) * N + nb + threadIdx.x];
    #pragma unroll
    for (int i = 0; i < 4; ++i)
        tile[threadIdx.y + i * 8][threadIdx.x] = r[i];
    __syncthreads();
    #pragma unroll
    for (int i = 0; i < 4; ++i)
        r[i] = tile[threadIdx.x][threadIdx.y + i * 8];
    #pragma unroll
    for (int i = 0; i < 4; ++i)
        out[(size_t)(nb + threadIdx.y + i * 8) * K + kb + threadIdx.x] = to_bf16(r[i]);
}

// ---------------------------------------------------------------------------
// WMMA GEMM:  C[M x N] = A[M x K](bf16) @ W + bias,  W given as WT[N x K](bf16)
// Block tile 128x128, 8 waves (4 x 2), wave tile 32x64 (2x4 WMMA 16x16x32).
// Staging: double-buffered global_load_async_to_lds_b128, slice loop unrolled
// by 2 so buffer indices are compile-time constants (no cndmask selects).
// M % 128 == 0, N % 128 == 0, K % 64 == 0 (nsl even).
// ---------------------------------------------------------------------------
#define TM 128
#define TN 128
#define TK 32

__global__ __launch_bounds__(256)
void gemm_bf16_wmma_kernel(const bf16_t* __restrict__ A,
                           const bf16_t* __restrict__ WT,
                           const float*  __restrict__ bias,
                           float* __restrict__ C,
                           int M, int N, int K) {
    __shared__ bf16_t Alds[2][TM * TK];   // 2 x 8 KB
    __shared__ bf16_t Blds[2][TN * TK];   // 2 x 8 KB

    const int tid   = threadIdx.x;
    const int lane  = tid & 31;
    const int wid   = tid >> 5;
    const int waveM = wid & 3;     // 4 wave-rows of 32
    const int waveN = wid >> 2;    // 2 wave-cols of 64
    const int bm = blockIdx.x * TM;
    const int bn = blockIdx.y * TN;

    // Staging map: 128x32 bf16 tile = 512 x 16B chunks; thread handles chunks
    // tid and tid+256 -> rows row0 and row0+64, 16B sub-chunk c4 within 64B row.
    const int row0 = tid >> 2;   // 0..63
    const int c4   = tid & 3;    // 0..3
    unsigned voffA[2], voffB[2];
    unsigned ldsA[2][2], ldsB[2][2];   // [chunk][buffer]
    #pragma unroll
    for (int i = 0; i < 2; ++i) {
        int r = row0 + 64 * i;
        voffA[i] = (unsigned)(((size_t)(bm + r) * K + c4 * 8) * 2);
        voffB[i] = (unsigned)(((size_t)(bn + r) * K + c4 * 8) * 2);
        #pragma unroll
        for (int s = 0; s < 2; ++s) {
            ldsA[i][s] = lds_off_u32(&Alds[s][r * TK + c4 * 8]);
            ldsB[i][s] = lds_off_u32(&Blds[s][r * TK + c4 * 8]);
        }
    }

    v8f acc[2][4];
    #pragma unroll
    for (int i = 0; i < 2; ++i)
        #pragma unroll
        for (int j = 0; j < 4; ++j)
            #pragma unroll
            for (int r = 0; r < 8; ++r) acc[i][j][r] = 0.0f;

    const int mrow = lane & 15;
    const int hi   = lane >> 4;
    const int nsl  = K / TK;      // even (24 or 96)

    // issue 4 async b128 copies for the next slice into buffer `buf`,
    // then advance the running global offsets by one K-slice.
    auto issue = [&](int buf) {
        #pragma unroll
        for (int i = 0; i < 2; ++i) {
            async_copy_b128(ldsA[i][buf], A,  voffA[i]);
            async_copy_b128(ldsB[i][buf], WT, voffB[i]);
            voffA[i] += TK * 2;
            voffB[i] += TK * 2;
        }
    };

    auto compute = [&](int buf) {
        const bf16_t* Ac = Alds[buf];
        const bf16_t* Bc = Blds[buf];
        // A 16x32 frag: lanes 0-15 hold K 0..7 / 16..23; lanes 16-31 K 8..15 / 24..31
        v16bf afrag[2];
        #pragma unroll
        for (int i = 0; i < 2; ++i) {
            const bf16_t* ap = Ac + (waveM * 32 + i * 16 + mrow) * TK;
            int kb = hi * 8;
            #pragma unroll
            for (int j = 0; j < 8; ++j) {
                afrag[i][j]     = ap[kb + j];
                afrag[i][8 + j] = ap[16 + kb + j];
            }
        }
        // B 32x16 frag: lanes 0-15 = col n with K 0..7 / 16..23; lanes 16-31 K 8..15 / 24..31
        v16bf bfrag[4];
        #pragma unroll
        for (int j2 = 0; j2 < 4; ++j2) {
            const bf16_t* bp = Bc + (waveN * 64 + j2 * 16 + mrow) * TK;
            #pragma unroll
            for (int j = 0; j < 8; ++j) {
                bfrag[j2][j]     = bp[hi * 8 + j];
                bfrag[j2][8 + j] = bp[16 + hi * 8 + j];
            }
        }
        #pragma unroll
        for (int i = 0; i < 2; ++i)
            #pragma unroll
            for (int j2 = 0; j2 < 4; ++j2)
                acc[i][j2] = __builtin_amdgcn_wmma_f32_16x16x32_bf16(
                    false, afrag[i], false, bfrag[j2], (short)0, acc[i][j2], false, false);
    };

    // prologue: slice 0 -> buffer 0
    issue(0);
    // main loop: two slices per iteration, constant buffer indices
    for (int s = 0; s < nsl - 2; s += 2) {
        issue(1); wait_asynccnt<4>(); __syncthreads(); compute(0); __syncthreads();
        issue(0); wait_asynccnt<4>(); __syncthreads(); compute(1); __syncthreads();
    }
    // epilogue: last two slices
    issue(1); wait_asynccnt<4>(); __syncthreads(); compute(0); __syncthreads();
    wait_asynccnt<0>(); __syncthreads(); compute(1);

    // store: lane<16 -> rows 0..7 / col=lane ; lane>=16 -> rows 8..15 / col=lane-16
    const int col   = lane & 15;
    const int rbase = hi * 8;
    #pragma unroll
    for (int i = 0; i < 2; ++i)
        #pragma unroll
        for (int j2 = 0; j2 < 4; ++j2) {
            int cg = bn + waveN * 64 + j2 * 16 + col;
            float bv = bias ? bias[cg] : 0.0f;
            #pragma unroll
            for (int r = 0; r < 8; ++r) {
                int rg = bm + waveM * 32 + i * 16 + rbase + r;
                C[(size_t)rg * N + cg] = acc[i][j2][r] + bv;
            }
        }
}

// ---------------------------------------------------------------------------
// Block-wide mean/var reduction helper (256 threads)
// ---------------------------------------------------------------------------
__device__ __forceinline__ void block_meanvar(float s1, float s2,
                                              float* r1, float* r2,
                                              float& mean, float& inv) {
    r1[threadIdx.x] = s1; r2[threadIdx.x] = s2;
    __syncthreads();
    for (int st = 128; st > 0; st >>= 1) {
        if (threadIdx.x < st) {
            r1[threadIdx.x] += r1[threadIdx.x + st];
            r2[threadIdx.x] += r2[threadIdx.x + st];
        }
        __syncthreads();
    }
    mean = r1[0] * (1.0f / H_);
    float var = r2[0] * (1.0f / H_) - mean * mean;
    inv = rsqrtf(var + 1e-5f);
}

// ---------------------------------------------------------------------------
// Embedding + LayerNorm: one block per token
// ---------------------------------------------------------------------------
__global__ __launch_bounds__(256)
void embed_ln_kernel(const int* __restrict__ ids,
                     const float* __restrict__ wemb, const float* __restrict__ pemb,
                     const float* __restrict__ temb,
                     const float* __restrict__ g, const float* __restrict__ bta,
                     float* __restrict__ x) {
    __shared__ float r1[256], r2[256];
    int tok = blockIdx.x;
    int s   = tok % S_;
    int id  = ids[tok];
    float e[3], s1 = 0.f, s2 = 0.f;
    #pragma unroll
    for (int j = 0; j < 3; ++j) {
        int c = threadIdx.x + 256 * j;
        float v = wemb[(size_t)id * H_ + c] + pemb[(size_t)(s + 2) * H_ + c] + temb[c];
        e[j] = v; s1 += v; s2 += v * v;
    }
    float mean, inv;
    block_meanvar(s1, s2, r1, r2, mean, inv);
    #pragma unroll
    for (int j = 0; j < 3; ++j) {
        int c = threadIdx.x + 256 * j;
        x[(size_t)tok * H_ + c] = (e[j] - mean) * inv * g[c] + bta[c];
    }
}

// ---------------------------------------------------------------------------
// Residual add + LayerNorm (in place on x): one block per token
// ---------------------------------------------------------------------------
__global__ __launch_bounds__(256)
void add_ln_kernel(float* __restrict__ x, const float* __restrict__ t,
                   const float* __restrict__ g, const float* __restrict__ bta) {
    __shared__ float r1[256], r2[256];
    int tok = blockIdx.x;
    float e[3], s1 = 0.f, s2 = 0.f;
    #pragma unroll
    for (int j = 0; j < 3; ++j) {
        int c = threadIdx.x + 256 * j;
        float v = x[(size_t)tok * H_ + c] + t[(size_t)tok * H_ + c];
        e[j] = v; s1 += v; s2 += v * v;
    }
    float mean, inv;
    block_meanvar(s1, s2, r1, r2, mean, inv);
    #pragma unroll
    for (int j = 0; j < 3; ++j) {
        int c = threadIdx.x + 256 * j;
        x[(size_t)tok * H_ + c] = (e[j] - mean) * inv * g[c] + bta[c];
    }
}

// ---------------------------------------------------------------------------
// Exact GELU elementwise, 4 elements/thread
// ---------------------------------------------------------------------------
__global__ __launch_bounds__(256)
void gelu_kernel(float* __restrict__ h, int n4) {
    int i = blockIdx.x * 256 + threadIdx.x;
    if (i < n4) {
        v4f v = reinterpret_cast<const v4f*>(h)[i];
        #pragma unroll
        for (int j = 0; j < 4; ++j)
            v[j] = 0.5f * v[j] * (1.0f + erff(v[j] * 0.70710678118654752f));
        reinterpret_cast<v4f*>(h)[i] = v;
    }
}

// ---------------------------------------------------------------------------
// Flash-style attention: one block per (b, head); one thread per query row.
// K/V tiles staged via async b128 DMA into LDS; online softmax.
// ---------------------------------------------------------------------------
__global__ __launch_bounds__(256)
void attention_kernel(const float* __restrict__ Q, const float* __restrict__ Kb,
                      const float* __restrict__ Vb, const int* __restrict__ amask,
                      float* __restrict__ ctx) {
    __shared__ float Kl[64 * 64];
    __shared__ float Vl[64 * 64];
    int bh = blockIdx.x;
    int b  = bh / NH_;
    int h  = bh % NH_;
    int qr = threadIdx.x;
    size_t rowbase = ((size_t)b * S_ + qr) * H_ + h * DH_;

    // staging map: tile 64x64 f32 = 1024 x 16B chunks; 4 per thread
    unsigned ldsK[4], ldsV[4], voff[4];
    #pragma unroll
    for (int i = 0; i < 4; ++i) {
        int li = threadIdx.x + i * 256;
        int kk = li >> 4, c = li & 15;
        ldsK[i] = lds_off_u32((const char*)Kl + li * 16);
        ldsV[i] = lds_off_u32((const char*)Vl + li * 16);
        voff[i] = (unsigned)((((size_t)b * S_ + kk) * H_ + h * DH_ + c * 4) * 4);
    }
    const unsigned kstep = (unsigned)(64 * H_ * 4);

    float q[DH_], acc[DH_];
    #pragma unroll
    for (int j = 0; j < DH_; ++j) { q[j] = Q[rowbase + j]; acc[j] = 0.f; }
    float m = -3.4e38f, l = 0.f;

    for (int kt = 0; kt < 4; ++kt) {
        __syncthreads();               // previous tile fully consumed
        #pragma unroll
        for (int i = 0; i < 4; ++i) {
            async_copy_b128(ldsK[i], Kb, voff[i] + kt * kstep);
            async_copy_b128(ldsV[i], Vb, voff[i] + kt * kstep);
        }
        wait_asynccnt<0>();
        __syncthreads();
        for (int kk = 0; kk < 64; ++kk) {
            int kpos = kt * 64 + kk;
            float sc = 0.f;
            #pragma unroll
            for (int j = 0; j < DH_; ++j) sc += q[j] * Kl[kk * 64 + j];
            sc = sc * 0.125f + (amask[b * S_ + kpos] ? 0.f : -1e9f);
            if (sc > m) {
                float al = __expf(m - sc);
                #pragma unroll
                for (int j = 0; j < DH_; ++j) acc[j] *= al;
                l *= al; m = sc;
            }
            float p = __expf(sc - m);
            l += p;
            #pragma unroll
            for (int j = 0; j < DH_; ++j) acc[j] += p * Vl[kk * 64 + j];
        }
    }
    float inv = 1.f / l;
    #pragma unroll
    for (int j = 0; j < DH_; ++j) ctx[rowbase + j] = acc[j] * inv;
}

// ---------------------------------------------------------------------------
// Word gather: positions of first-piece labels, in order (stable-argsort equiv.)
// ---------------------------------------------------------------------------
__global__ void build_word_index_kernel(const int* __restrict__ labels,
                                        int* __restrict__ widx, int* __restrict__ wcount) {
    int b = threadIdx.x;
    if (b >= B_) return;
    int cnt = 0;
    for (int s = 0; s < S_ && cnt < W_; ++s)
        if (labels[b * S_ + s] != IGNORE_IDX) widx[b * W_ + cnt++] = s;
    wcount[b] = cnt;
    for (int i = cnt; i < W_; ++i) widx[b * W_ + i] = 0;
}

// ---------------------------------------------------------------------------
// Emissions: em[b,w,t] = hidden[b, widx[b,w]] . cls_w[:,t] + cls_b[t]
// ---------------------------------------------------------------------------
__global__ __launch_bounds__(256)
void emissions_kernel(const float* __restrict__ hidden, const int* __restrict__ widx,
                      const int* __restrict__ wcount, const float* __restrict__ cw,
                      const float* __restrict__ cb, float* __restrict__ em) {
    int idx = blockIdx.x * 256 + threadIdx.x;
    if (idx >= B_ * W_ * T_) return;
    int t = idx % T_;
    int w = (idx / T_) % W_;
    int b = idx / (T_ * W_);
    float r = cb[t];
    if (w < wcount[b]) {
        const float* hrow = hidden + ((size_t)b * S_ + widx[b * W_ + w]) * H_;
        float s = 0.f;
        for (int i = 0; i < H_; ++i) s += hrow[i] * cw[i * T_ + t];
        r += s;
    }
    em[idx] = r;
}

// ---------------------------------------------------------------------------
// CRF NLL: one thread per batch, serial forward algorithm (tiny: W=128, T=9)
// ---------------------------------------------------------------------------
__global__ void crf_kernel(const float* __restrict__ em, const int* __restrict__ tags,
                           const int* __restrict__ wcount,
                           const float* __restrict__ start, const float* __restrict__ endv,
                           const float* __restrict__ trans, float* __restrict__ out) {
    __shared__ float llh[B_];
    int b = threadIdx.x;
    if (b < B_) {
        const float* e = em + (size_t)b * W_ * T_;
        const int* tg = tags + b * W_;
        int cnt = wcount[b];
        float num = start[tg[0]] + e[tg[0]];
        int prev = tg[0];
        for (int w = 1; w < W_; ++w) {
            if (w < cnt) {
                num += trans[prev * T_ + tg[w]] + e[w * T_ + tg[w]];
                prev = tg[w];
            }
        }
        num += endv[prev];
        float alpha[T_], an[T_];
        for (int t = 0; t < T_; ++t) alpha[t] = start[t] + e[t];
        for (int w = 1; w < W_; ++w) {
            if (w < cnt) {
                for (int t = 0; t < T_; ++t) {
                    float mx = -3.4e38f;
                    for (int u = 0; u < T_; ++u)
                        mx = fmaxf(mx, alpha[u] + trans[u * T_ + t]);
                    float sden = 0.f;
                    for (int u = 0; u < T_; ++u)
                        sden += __expf(alpha[u] + trans[u * T_ + t] - mx);
                    an[t] = mx + __logf(sden) + e[w * T_ + t];
                }
                for (int t = 0; t < T_; ++t) alpha[t] = an[t];
            }
        }
        float mx = -3.4e38f;
        for (int t = 0; t < T_; ++t) mx = fmaxf(mx, alpha[t] + endv[t]);
        float sden = 0.f;
        for (int t = 0; t < T_; ++t) sden += __expf(alpha[t] + endv[t] - mx);
        llh[b] = num - (mx + __logf(sden));
    }
    __syncthreads();
    if (threadIdx.x == 0) {
        float s = 0.f;
        for (int i = 0; i < B_; ++i) s += llh[i];
        out[0] = -s / (float)B_;
    }
}

// ---------------------------------------------------------------------------
// Host orchestration
// ---------------------------------------------------------------------------
extern "C" void kernel_launch(void* const* d_in, const int* in_sizes, int n_in,
                              void* d_out, int out_size, void* d_ws, size_t ws_size,
                              hipStream_t stream) {
    const float* word_emb  = (const float*)d_in[0];
    const float* pos_emb   = (const float*)d_in[1];
    const float* type_emb  = (const float*)d_in[2];
    const float* emb_ln_s  = (const float*)d_in[3];
    const float* emb_ln_b  = (const float*)d_in[4];
    const float* qw        = (const float*)d_in[5];
    const float* qb        = (const float*)d_in[6];
    const float* kw        = (const float*)d_in[7];
    const float* kb        = (const float*)d_in[8];
    const float* vw        = (const float*)d_in[9];
    const float* vb        = (const float*)d_in[10];
    const float* ow        = (const float*)d_in[11];
    const float* ob        = (const float*)d_in[12];
    const float* attn_ln_s = (const float*)d_in[13];
    const float* attn_ln_b = (const float*)d_in[14];
    const float* w1        = (const float*)d_in[15];
    const float* b1        = (const float*)d_in[16];
    const float* w2        = (const float*)d_in[17];
    const float* b2        = (const float*)d_in[18];
    const float* ffn_ln_s  = (const float*)d_in[19];
    const float* ffn_ln_b  = (const float*)d_in[20];
    const float* cls_w     = (const float*)d_in[21];
    const float* cls_b     = (const float*)d_in[22];
    const float* crf_start = (const float*)d_in[23];
    const float* crf_end   = (const float*)d_in[24];
    const float* crf_trans = (const float*)d_in[25];
    const int*   input_ids = (const int*)d_in[26];
    const int*   amask     = (const int*)d_in[27];
    const int*   label_ids = (const int*)d_in[28];
    const int*   tag_ids   = (const int*)d_in[29];
    // d_in[30] word_lengths: encoded in label_ids, unused

    char* ws = (char*)d_ws;
    size_t off = 0;
    auto alloc = [&](size_t bytes) -> void* {
        void* p = ws + off;
        off += (bytes + 255) & ~(size_t)255;
        return p;
    };
    float*  x    = (float*) alloc((size_t)NTOK * H_ * 4);
    float*  qf   = (float*) alloc((size_t)NTOK * H_ * 4);
    float*  kf   = (float*) alloc((size_t)NTOK * H_ * 4);
    float*  vf   = (float*) alloc((size_t)NTOK * H_ * 4);
    float*  ctx  = (float*) alloc((size_t)NTOK * H_ * 4);
    float*  t1   = (float*) alloc((size_t)NTOK * H_ * 4);
    float*  hffn = (float*) alloc((size_t)NTOK * F_ * 4);
    bf16_t* abf  = (bf16_t*)alloc((size_t)NTOK * F_ * 2);
    bf16_t* wt   = (bf16_t*)alloc((size_t)F_ * H_ * 2);
    int*    widx = (int*)   alloc((size_t)B_ * W_ * 4);
    int*    wcnt = (int*)   alloc((size_t)B_ * 4);
    float*  em   = (float*) alloc((size_t)B_ * W_ * T_ * 4);
    (void)ws_size; (void)in_sizes; (void)n_in; (void)out_size;

    const dim3 blk256(256);
    const dim3 tblk(32, 8);

    embed_ln_kernel<<<NTOK, blk256, 0, stream>>>(input_ids, word_emb, pos_emb,
                                                 type_emb, emb_ln_s, emb_ln_b, x);

    for (int l = 0; l < L_; ++l) {
        const float* qw_l = qw + (size_t)l * H_ * H_;
        const float* kw_l = kw + (size_t)l * H_ * H_;
        const float* vw_l = vw + (size_t)l * H_ * H_;
        const float* ow_l = ow + (size_t)l * H_ * H_;
        const float* w1_l = w1 + (size_t)l * H_ * F_;
        const float* w2_l = w2 + (size_t)l * F_ * H_;

        cvt_bf16_kernel<<<(NTOK * H_ / 4 + 255) / 256, blk256, 0, stream>>>(x, abf, NTOK * H_ / 4);

        transpose_bf16_kernel<<<dim3(H_ / 32, H_ / 32), tblk, 0, stream>>>(qw_l, wt, H_, H_);
        gemm_bf16_wmma_kernel<<<dim3(NTOK / TM, H_ / TN), blk256, 0, stream>>>(
            abf, wt, qb + (size_t)l * H_, qf, NTOK, H_, H_);
        transpose_bf16_kernel<<<dim3(H_ / 32, H_ / 32), tblk, 0, stream>>>(kw_l, wt, H_, H_);
        gemm_bf16_wmma_kernel<<<dim3(NTOK / TM, H_ / TN), blk256, 0, stream>>>(
            abf, wt, kb + (size_t)l * H_, kf, NTOK, H_, H_);
        transpose_bf16_kernel<<<dim3(H_ / 32, H_ / 32), tblk, 0, stream>>>(vw_l, wt, H_, H_);
        gemm_bf16_wmma_kernel<<<dim3(NTOK / TM, H_ / TN), blk256, 0, stream>>>(
            abf, wt, vb + (size_t)l * H_, vf, NTOK, H_, H_);

        attention_kernel<<<B_ * NH_, blk256, 0, stream>>>(qf, kf, vf, amask, ctx);

        cvt_bf16_kernel<<<(NTOK * H_ / 4 + 255) / 256, blk256, 0, stream>>>(ctx, abf, NTOK * H_ / 4);
        transpose_bf16_kernel<<<dim3(H_ / 32, H_ / 32), tblk, 0, stream>>>(ow_l, wt, H_, H_);
        gemm_bf16_wmma_kernel<<<dim3(NTOK / TM, H_ / TN), blk256, 0, stream>>>(
            abf, wt, ob + (size_t)l * H_, t1, NTOK, H_, H_);
        add_ln_kernel<<<NTOK, blk256, 0, stream>>>(x, t1, attn_ln_s + (size_t)l * H_,
                                                   attn_ln_b + (size_t)l * H_);

        cvt_bf16_kernel<<<(NTOK * H_ / 4 + 255) / 256, blk256, 0, stream>>>(x, abf, NTOK * H_ / 4);
        transpose_bf16_kernel<<<dim3(H_ / 32, F_ / 32), tblk, 0, stream>>>(w1_l, wt, H_, F_);
        gemm_bf16_wmma_kernel<<<dim3(NTOK / TM, F_ / TN), blk256, 0, stream>>>(
            abf, wt, b1 + (size_t)l * F_, hffn, NTOK, F_, H_);
        gelu_kernel<<<(NTOK * F_ / 4 + 255) / 256, blk256, 0, stream>>>(hffn, NTOK * F_ / 4);

        cvt_bf16_kernel<<<(NTOK * F_ / 4 + 255) / 256, blk256, 0, stream>>>(hffn, abf, NTOK * F_ / 4);
        transpose_bf16_kernel<<<dim3(F_ / 32, H_ / 32), tblk, 0, stream>>>(w2_l, wt, F_, H_);
        gemm_bf16_wmma_kernel<<<dim3(NTOK / TM, H_ / TN), blk256, 0, stream>>>(
            abf, wt, b2 + (size_t)l * H_, t1, NTOK, H_, F_);
        add_ln_kernel<<<NTOK, blk256, 0, stream>>>(x, t1, ffn_ln_s + (size_t)l * H_,
                                                   ffn_ln_b + (size_t)l * H_);
    }

    build_word_index_kernel<<<1, 32, 0, stream>>>(label_ids, widx, wcnt);
    emissions_kernel<<<(B_ * W_ * T_ + 255) / 256, blk256, 0, stream>>>(
        x, widx, wcnt, cls_w, cls_b, em);
    crf_kernel<<<1, 32, 0, stream>>>(em, tag_ids, wcnt, crf_start, crf_end, crf_trans,
                                     (float*)d_out);
}
